// SlotAttention_75093208203460
// MI455X (gfx1250) — compile-verified, matrix operand order checked by reading, not run on hardware
//
// Slot Attention for MI455X (gfx1250), wave32 + WMMA bf16.
// Precision/data-movement rationale:
//  - k,v stored bf16 -> 134 MB combined, fits the 192 MB L2, so the 3
//    attention iterations never touch HBM for k/v.
//  - All long-reduction matmuls use v_wmma_f32_16x16x32_bf16 (f32 accum).
//  - HBM traffic ~0.5 GB (~20us @ 23.3 TB/s) vs ~43 GFLOP bf16 WMMA: balanced.
//  - bf16 conversion via native hardware cast (RNE), not SW bit-twiddling.
#include <hip/hip_runtime.h>
#include <hip/hip_bf16.h>
#include <math.h>

#define BB 32
#define NNT 4096
#define DD 256
#define SS 16
#define HHD 512

typedef __attribute__((ext_vector_type(16))) __bf16 v16bf;
typedef __attribute__((ext_vector_type(8)))  float  v8f;

__device__ __forceinline__ unsigned short f2bf(float f) {
  return __builtin_bit_cast(unsigned short, (__bf16)f);  // hw RNE cvt
}

// Load a 16x32 bf16 fragment from a [16][stride] row-major tile.
// Works for A ([M][K]) and for B when the tile is stored as [N][K].
// ISA 7.12.2: lanes 0-15 = rows, K 0..7/16..23; lanes 16-31 same rows, K 8..15/24..31.
// Layout chosen so the 8 dword loads fuse into 2x b128 loads (verified in disasm).
__device__ __forceinline__ v16bf frag16(const unsigned short* base, int stride, int lane) {
  const int half = (lane >> 4) & 1;
  const unsigned short* row = base + (lane & 15) * stride;
  v16bf out;
#pragma unroll
  for (int p = 0; p < 8; ++p) {
    int k = (p < 4) ? (half * 8 + p * 2) : (16 + half * 8 + (p - 4) * 2);
    unsigned w = *(const unsigned*)(row + k);
    out[2 * p]     = __builtin_bit_cast(__bf16, (unsigned short)(w & 0xFFFFu));
    out[2 * p + 1] = __builtin_bit_cast(__bf16, (unsigned short)(w >> 16));
  }
  return out;
}

__device__ __forceinline__ v8f wmma_bf(v16bf a, v16bf b, v8f c) {
  return __builtin_amdgcn_wmma_f32_16x16x32_bf16(false, a, false, b, (short)0, c,
                                                 false, false);
}

// ---------- small elementwise kernels ----------
__global__ void k_cvt(const float* __restrict__ s, unsigned short* __restrict__ d, int n) {
  int i = blockIdx.x * 256 + threadIdx.x;
  if (i < n) d[i] = f2bf(s[i]);
}

__global__ void k_zero(float* p, int n) {
  int i = blockIdx.x * 256 + threadIdx.x;
  if (i < n) p[i] = 0.f;
}

__global__ void k_sinit(const float* __restrict__ mu, const float* __restrict__ ls,
                        const float* __restrict__ noise, float* __restrict__ slots) {
  int i = blockIdx.x * 256 + threadIdx.x;  // B*S*D = 131072
  int d = i & (DD - 1);
  float sg = fmaxf(expf(ls[d]), 1e-6f);
  slots[i] = mu[d] + sg * noise[i];
}

__global__ void k_gru(const float* __restrict__ gi, const float* __restrict__ gh,
                      const float* __restrict__ hprev, float* __restrict__ slots) {
  int i = blockIdx.x * 256 + threadIdx.x;  // 512*256
  int row = i >> 8, c = i & 255;
  const float* gir = gi + row * 768;
  const float* ghr = gh + row * 768;
  float r = 1.f / (1.f + expf(-(gir[c] + ghr[c])));
  float z = 1.f / (1.f + expf(-(gir[256 + c] + ghr[256 + c])));
  float nn = tanhf(gir[512 + c] + r * ghr[512 + c]);
  float h = hprev[i];
  slots[i] = (1.f - z) * nn + z * h;
}

__global__ void k_anorm(const float* __restrict__ au, const float* __restrict__ cs,
                        float* __restrict__ out) {
  long i = (long)blockIdx.x * 256 + threadIdx.x;  // B*N*S
  int s = (int)(i & 15);
  int b = (int)(i >> 16);  // i>>4 = token idx, >>12 more = batch
  out[i] = au[i] / (cs[b * SS + s] + 1e-8f);
}

__global__ void __launch_bounds__(256) k_ln_rows(const float* __restrict__ in,
                                                 const float* __restrict__ w,
                                                 const float* __restrict__ b,
                                                 float* __restrict__ out) {
  __shared__ float sb[256], sb2[256];
  int row = blockIdx.x, t = threadIdx.x;
  float v = in[row * DD + t];
  sb[t] = v; sb2[t] = v * v;
  __syncthreads();
  for (int s = 128; s > 0; s >>= 1) {
    if (t < s) { sb[t] += sb[t + s]; sb2[t] += sb2[t + s]; }
    __syncthreads();
  }
  float m = sb[0] * (1.f / 256.f);
  float var = sb2[0] * (1.f / 256.f) - m * m;
  float ir = rsqrtf(var + 1e-5f);
  out[row * DD + t] = (v - m) * ir * w[t] + b[t];
}

// ---------- fused LayerNorm + K/V projection (bf16 out) ----------
// 64 rows/block, 8 waves. 1024 wmma issues per block.
__global__ void __launch_bounds__(256) k_ln_kv(
    const float* __restrict__ x, const float* __restrict__ lw,
    const float* __restrict__ lb, const unsigned short* __restrict__ wkbf,
    const unsigned short* __restrict__ wvbf, unsigned short* __restrict__ kb,
    unsigned short* __restrict__ vb) {
  __shared__ unsigned short xs[64 * 264];
  __shared__ float ps[64][4], ps2[64][4], mu[64], rs[64];
  const int tid = threadIdx.x;
  const long rowg = (long)blockIdx.x * 64;
  const int r = tid >> 2, q = tid & 3;
  {
    const float* xr = x + (rowg + r) * DD + q * 64;
    float s = 0.f, s2 = 0.f;
    for (int j = 0; j < 64; ++j) { float v = xr[j]; s += v; s2 += v * v; }
    ps[r][q] = s; ps2[r][q] = s2;
  }
  __syncthreads();
  if (q == 0) {
    float s = ps[r][0] + ps[r][1] + ps[r][2] + ps[r][3];
    float s2 = ps2[r][0] + ps2[r][1] + ps2[r][2] + ps2[r][3];
    float m = s * (1.f / 256.f);
    float v = s2 * (1.f / 256.f) - m * m;
    mu[r] = m; rs[r] = rsqrtf(v + 1e-5f);
  }
  __syncthreads();
  {
    const float* xr = x + (rowg + r) * DD + q * 64;
    float m = mu[r], ir = rs[r];
    for (int j = 0; j < 64; ++j) {
      int c = q * 64 + j;
      xs[r * 264 + c] = f2bf((xr[j] - m) * ir * lw[c] + lb[c]);
    }
  }
  __syncthreads();
  const int wave = tid >> 5, lane = tid & 31;
  const int n = lane & 15, mh = (lane >> 4) * 8;
  for (int tc = wave; tc < 32; tc += 8) {
    const int wsel = tc >> 4, nt = tc & 15;
    const unsigned short* wb = (wsel ? wvbf : wkbf) + nt * 16 * DD;
    unsigned short* outp = wsel ? vb : kb;
    v8f a0 = {0,0,0,0,0,0,0,0}, a1 = a0, a2 = a0, a3 = a0;
    for (int kk = 0; kk < 8; ++kk) {
      v16bf bf = frag16(wb + kk * 32, DD, lane);
      v16bf f0 = frag16(xs + 0 * 16 * 264 + kk * 32, 264, lane);
      v16bf f1 = frag16(xs + 1 * 16 * 264 + kk * 32, 264, lane);
      v16bf f2 = frag16(xs + 2 * 16 * 264 + kk * 32, 264, lane);
      v16bf f3 = frag16(xs + 3 * 16 * 264 + kk * 32, 264, lane);
      a0 = wmma_bf(f0, bf, a0);
      a1 = wmma_bf(f1, bf, a1);
      a2 = wmma_bf(f2, bf, a2);
      a3 = wmma_bf(f3, bf, a3);
    }
    const int col = nt * 16 + n;
#pragma unroll
    for (int rr = 0; rr < 8; ++rr) {
      outp[(rowg + 0 * 16 + mh + rr) * DD + col] = f2bf(a0[rr]);
      outp[(rowg + 1 * 16 + mh + rr) * DD + col] = f2bf(a1[rr]);
      outp[(rowg + 2 * 16 + mh + rr) * DD + col] = f2bf(a2[rr]);
      outp[(rowg + 3 * 16 + mh + rr) * DD + col] = f2bf(a3[rr]);
    }
  }
}

// ---------- generic 16-row WMMA GEMM: out = act(A @ W^T + bias) (+ resid) ----------
__global__ void __launch_bounds__(256) k_gemm16(
    const float* __restrict__ A, const unsigned short* __restrict__ W,
    const float* __restrict__ bias, const float* __restrict__ resid,
    float* __restrict__ outf, unsigned short* __restrict__ outb, int K, int Nn,
    int act) {
  __shared__ unsigned short at[16 * 520];
  const int tid = threadIdx.x;
  const int m0 = blockIdx.x * 16;
  const int stride = K + 8;
  for (int idx = tid; idx < 16 * K; idx += 256) {
    int rr = idx / K, cc = idx - rr * K;
    at[rr * stride + cc] = f2bf(A[(m0 + rr) * K + cc]);
  }
  __syncthreads();
  const int wave = tid >> 5, lane = tid & 31;
  const int n = lane & 15, mh = (lane >> 4) * 8;
  for (int nt = wave; nt < (Nn >> 4); nt += 8) {
    v8f acc = {0,0,0,0,0,0,0,0};
    const unsigned short* wb = W + nt * 16 * K;
    for (int kk = 0; kk < (K >> 5); ++kk) {
      v16bf bf = frag16(wb + kk * 32, K, lane);
      v16bf af = frag16(at + kk * 32, stride, lane);
      acc = wmma_bf(af, bf, acc);
    }
    int col = nt * 16 + n;
    float bc = bias ? bias[col] : 0.f;
#pragma unroll
    for (int rr = 0; rr < 8; ++rr) {
      int row = m0 + mh + rr;
      float v = acc[rr] + bc;
      if (act) v = 0.5f * v * (1.f + erff(v * 0.70710678118f));
      if (resid) v += resid[row * Nn + col];
      if (outf) outf[row * Nn + col] = v;
      if (outb) outb[row * Nn + col] = f2bf(v);
    }
  }
}

// ---------- attention pass A: logits (WMMA) + softmax(S) + tw + colsums ----------
__global__ void __launch_bounds__(256) k_attn(
    const unsigned short* __restrict__ kb, const unsigned short* __restrict__ qb,
    const float* __restrict__ tw, float* __restrict__ attn_u,
    float* __restrict__ colsum) {
  __shared__ float lt[8][16][17];
  __shared__ float colpart[16];
  const int tid = threadIdx.x, wave = tid >> 5, lane = tid & 31;
  const int b = blockIdx.x >> 5;    // 32 blocks per batch
  const int seg = blockIdx.x & 31;  // 128 token rows each
  const int nbase = seg * 128 + wave * 16;
  if (tid < 16) colpart[tid] = 0.f;
  const unsigned short* arow = kb + (long)(b * NNT + nbase) * DD;
  const unsigned short* brow = qb + b * SS * DD;  // [s][d] == B^T layout
  v8f acc = {0,0,0,0,0,0,0,0};
  for (int kk = 0; kk < 8; ++kk) {
    v16bf af = frag16(arow + kk * 32, DD, lane);
    v16bf bf = frag16(brow + kk * 32, DD, lane);
    acc = wmma_bf(af, bf, acc);
  }
  const int n = lane & 15, mh = (lane >> 4) * 8;
#pragma unroll
  for (int rr = 0; rr < 8; ++rr) lt[wave][mh + rr][n] = acc[rr] * 0.0625f;  // D^-0.5
  __syncthreads();
  if (tid < 128) {
    int w2 = tid >> 4, rl = tid & 15;
    int nrow = seg * 128 + w2 * 16 + rl;
    float* lr = lt[w2][rl];
    float mx = lr[0];
    for (int s = 1; s < 16; ++s) mx = fmaxf(mx, lr[s]);
    float e[16], sum = 0.f;
    for (int s = 0; s < 16; ++s) { e[s] = expf(lr[s] - mx); sum += e[s]; }
    float twv = fmaxf(tw[b * NNT + nrow], 0.f);
    float inv = twv / sum;
    float* op = attn_u + (long)(b * NNT + nrow) * SS;
    for (int s = 0; s < 16; ++s) {
      float a = e[s] * inv;
      op[s] = a;
      atomicAdd(&colpart[s], a);
    }
  }
  __syncthreads();
  if (tid < 16) atomicAdd(&colsum[b * SS + tid], colpart[tid]);
}

// ---------- attention pass B: updates = (attn^T @ v) / (colsum+eps), K=N=4096 ----------
// K-step widened to 64 (2 wmma per barrier pair, 64 steps).
__global__ void __launch_bounds__(256) k_updates(
    const float* __restrict__ attn_u, const unsigned short* __restrict__ vb,
    const float* __restrict__ colsum, float* __restrict__ upd) {
  __shared__ unsigned short at[16 * 66];
  __shared__ unsigned short vt[128 * 66];
  const int tid = threadIdx.x, wave = tid >> 5, lane = tid & 31;
  const int b = blockIdx.x >> 1;
  const int dh = (blockIdx.x & 1) * 128;
  v8f acc = {0,0,0,0,0,0,0,0};
  for (int step = 0; step < 64; ++step) {
    const int n0 = step * 64;
    if (step < 63 && lane == 0)  // global_prefetch_b8 for the next v slab
      __builtin_prefetch(vb + (long)(b * NNT + n0 + 64 + wave * 8) * DD + dh, 0, 1);
    for (int idx = tid; idx < 1024; idx += 256) {  // attn tile, transposed to [s][n]
      int s = idx >> 6, nn = idx & 63;
      at[s * 66 + nn] = f2bf(attn_u[(long)(b * NNT + n0 + nn) * SS + s]);
    }
    for (int idx = tid; idx < 8192; idx += 256) {  // v tile, transposed to [d][n]
      int nn = idx >> 7, dd = idx & 127;
      vt[dd * 66 + nn] = vb[(long)(b * NNT + n0 + nn) * DD + dh + dd];
    }
    __syncthreads();
    v16bf a0 = frag16(at, 66, lane);
    v16bf a1 = frag16(at + 32, 66, lane);
    v16bf b0 = frag16(vt + wave * 16 * 66, 66, lane);
    v16bf b1 = frag16(vt + wave * 16 * 66 + 32, 66, lane);
    acc = wmma_bf(a0, b0, acc);
    acc = wmma_bf(a1, b1, acc);
    __syncthreads();
  }
  const int n = lane & 15, mh = (lane >> 4) * 8;
#pragma unroll
  for (int rr = 0; rr < 8; ++rr) {
    int s = mh + rr;
    float sc = 1.f / (colsum[b * SS + s] + 1e-8f);
    upd[(b * SS + s) * DD + dh + wave * 16 + n] = acc[rr] * sc;
  }
}

extern "C" void kernel_launch(void* const* d_in, const int* in_sizes, int n_in,
                              void* d_out, int out_size, void* d_ws, size_t ws_size,
                              hipStream_t stream) {
  const float* inputs   = (const float*)d_in[0];
  const float* tw       = (const float*)d_in[1];
  const float* noise    = (const float*)d_in[2];
  const float* ln_in_w  = (const float*)d_in[3];
  const float* ln_in_b  = (const float*)d_in[4];
  const float* ln_s_w   = (const float*)d_in[5];
  const float* ln_s_b   = (const float*)d_in[6];
  const float* ln_m_w   = (const float*)d_in[7];
  const float* ln_m_b   = (const float*)d_in[8];
  const float* Wk       = (const float*)d_in[9];
  const float* Wv       = (const float*)d_in[10];
  const float* Wq       = (const float*)d_in[11];
  const float* gru_wih  = (const float*)d_in[12];
  const float* gru_whh  = (const float*)d_in[13];
  const float* gru_bih  = (const float*)d_in[14];
  const float* gru_bhh  = (const float*)d_in[15];
  const float* mlp_w1   = (const float*)d_in[16];
  const float* mlp_b1   = (const float*)d_in[17];
  const float* mlp_w2   = (const float*)d_in[18];
  const float* mlp_b2   = (const float*)d_in[19];
  const float* slots_mu = (const float*)d_in[20];
  const float* slots_ls = (const float*)d_in[21];

  char* p = (char*)d_ws;
  auto alloc = [&](size_t bytes) -> char* {
    char* r = p;
    p += (bytes + 255) & ~(size_t)255;
    return r;
  };
  unsigned short* kb     = (unsigned short*)alloc((size_t)BB * NNT * DD * 2);
  unsigned short* vb     = (unsigned short*)alloc((size_t)BB * NNT * DD * 2);
  float* attn_u          = (float*)alloc((size_t)BB * NNT * SS * 4);
  float* colsum          = (float*)alloc(512 * 4);
  float* slots           = (float*)alloc(131072 * 4);
  float* slots_prev      = (float*)alloc(131072 * 4);
  float* upd             = (float*)alloc(131072 * 4);
  float* xs              = (float*)alloc(131072 * 4);
  float* gi              = (float*)alloc(512 * 768 * 4);
  float* gh              = (float*)alloc(512 * 768 * 4);
  float* h1              = (float*)alloc(512 * 512 * 4);
  unsigned short* qb     = (unsigned short*)alloc(512 * 256 * 2);
  unsigned short* wk_bf  = (unsigned short*)alloc(65536 * 2);
  unsigned short* wv_bf  = (unsigned short*)alloc(65536 * 2);
  unsigned short* wq_bf  = (unsigned short*)alloc(65536 * 2);
  unsigned short* wih_bf = (unsigned short*)alloc(196608 * 2);
  unsigned short* whh_bf = (unsigned short*)alloc(196608 * 2);
  unsigned short* w1_bf  = (unsigned short*)alloc(131072 * 2);
  unsigned short* w2_bf  = (unsigned short*)alloc(131072 * 2);

  dim3 blk(256);
  k_cvt<<<256, blk, 0, stream>>>(Wk, wk_bf, 65536);
  k_cvt<<<256, blk, 0, stream>>>(Wv, wv_bf, 65536);
  k_cvt<<<256, blk, 0, stream>>>(Wq, wq_bf, 65536);
  k_cvt<<<768, blk, 0, stream>>>(gru_wih, wih_bf, 196608);
  k_cvt<<<768, blk, 0, stream>>>(gru_whh, whh_bf, 196608);
  k_cvt<<<512, blk, 0, stream>>>(mlp_w1, w1_bf, 131072);
  k_cvt<<<512, blk, 0, stream>>>(mlp_w2, w2_bf, 131072);
  k_sinit<<<512, blk, 0, stream>>>(slots_mu, slots_ls, noise, slots);
  k_ln_kv<<<BB * NNT / 64, blk, 0, stream>>>(inputs, ln_in_w, ln_in_b, wk_bf,
                                             wv_bf, kb, vb);
  for (int it = 0; it < 3; ++it) {
    hipMemcpyAsync(slots_prev, slots, 131072 * 4, hipMemcpyDeviceToDevice, stream);
    k_ln_rows<<<512, blk, 0, stream>>>(slots, ln_s_w, ln_s_b, xs);
    k_gemm16<<<32, blk, 0, stream>>>(xs, wq_bf, nullptr, nullptr, nullptr, qb,
                                     256, 256, 0);
    k_zero<<<2, blk, 0, stream>>>(colsum, 512);
    k_attn<<<1024, blk, 0, stream>>>(kb, qb, tw, attn_u, colsum);
    k_updates<<<64, blk, 0, stream>>>(attn_u, vb, colsum, upd);
    k_gemm16<<<32, blk, 0, stream>>>(upd, wih_bf, gru_bih, nullptr, gi, nullptr,
                                     256, 768, 0);
    k_gemm16<<<32, blk, 0, stream>>>(slots_prev, whh_bf, gru_bhh, nullptr, gh,
                                     nullptr, 256, 768, 0);
    k_gru<<<512, blk, 0, stream>>>(gi, gh, slots_prev, slots);
    k_ln_rows<<<512, blk, 0, stream>>>(slots, ln_m_w, ln_m_b, xs);
    k_gemm16<<<32, blk, 0, stream>>>(xs, w1_bf, mlp_b1, nullptr, h1, nullptr,
                                     256, 512, 1);
    k_gemm16<<<32, blk, 0, stream>>>(h1, w2_bf, mlp_b2, slots, slots, nullptr,
                                     512, 256, 0);
  }
  float* out = (float*)d_out;
  hipMemcpyAsync(out, slots, 131072 * 4, hipMemcpyDeviceToDevice, stream);
  k_anorm<<<8192, blk, 0, stream>>>(attn_u, colsum, out + BB * SS * DD);
}
